// LstmDecoder_57767310131741
// MI455X (gfx1250) — matrix operand order repaired and checked by reading, not compiled, not run on hardware
//
#include <hip/hip_runtime.h>
#include <hip/hip_bf16.h>
#include <stdint.h>

// ---------------------------------------------------------------------------
// Attention-LSTM decoder for MI455X (gfx1250, wave32, WMMA).
// All large GEMMs run through V_WMMA_F32_16X16X32_BF16 (bf16 in, f32 acc).
// Tile staging uses GLOBAL_LOAD_ASYNC_TO_LDS_B128 (ASYNCcnt) when available.
// ---------------------------------------------------------------------------

typedef __attribute__((ext_vector_type(16))) __bf16 v16bf;
typedef __attribute__((ext_vector_type(8)))  float  v8f;

#define BDIM 128
#define B_   128
#define ENC_ 2048
#define HW_  196
#define E_   512
#define D_   512
#define V_   10000
#define T_   32

#if __has_builtin(__builtin_amdgcn_global_load_async_to_lds_b128) && \
    __has_builtin(__builtin_amdgcn_s_wait_asynccnt)
#define HAS_ASYNC_LDS 1
#else
#define HAS_ASYNC_LDS 0
#endif

// Exact parameter types of the async-LDS builtin (from hipcc diagnostics):
// pointer to 16-byte int vector, in global (AS1) / LDS (AS3) address spaces.
typedef int v4i_ __attribute__((__vector_size__(16)));
typedef __attribute__((address_space(1))) v4i_ g_v4i;
typedef __attribute__((address_space(3))) v4i_ l_v4i;

__device__ __forceinline__ float bf2f(unsigned short u) {
  return __uint_as_float(((unsigned)u) << 16);
}
__device__ __forceinline__ unsigned short f2bf(float f) {
  unsigned u = __float_as_uint(f);
  u += 0x7FFFu + ((u >> 16) & 1u);          // round-to-nearest-even
  return (unsigned short)(u >> 16);
}

// -------------------------- generic WMMA GEMM ------------------------------
// D[M x N] = act( sum_s A_s[M x K_s] @ B_s[N x K_s]^T + bias0 + bias1 )
// A, B bf16 row-major. Optional "x-mode": segment 0's A is read from the f32
// tensor x with an on-the-fly transpose (A[m][k] = x[k*xRows + m]).
struct Seg {
  const unsigned short* A;
  const unsigned short* Bw;
  long long aBatch;     // element stride of A per blockIdx.z
  int lda, ldb, K;      // K must be a multiple of 64
};

struct GemmP {
  Seg seg[3];
  int nseg;
  const float* bias0;
  const float* bias1;
  float* outF;                 // optional f32 output
  unsigned short* outB;        // optional bf16 output
  long long oBatch;            // element stride of out per blockIdx.z
  int ldc;
  int N;
  int Mstore;                  // rows actually written (handles ragged M)
  int act;                     // 0 = none, 1 = tanh
  const float* xsrc;           // x-mode source (f32)
  long long xBatch;            // element stride per blockIdx.z
  int xRows;                   // inner stride / row clamp (= HW)
  int aXmode;
};

// LDS row stride: 36 dwords = 144 bytes (16B aligned, bank-spreading).
#define ARS 36
#define BRS 36

__global__ __launch_bounds__(BDIM)
void wmma_gemm_kernel(GemmP p) {
  const int tid  = threadIdx.x;
  const int lane = tid & 31;
  const int wave = tid >> 5;
  const int n0   = blockIdx.x * 128 + wave * 32;   // wave owns 16x32 output
  const int m0   = blockIdx.y * 16;
  const int bz   = blockIdx.z;
  const bool act0 = (n0 < p.N);
  const bool act1 = (n0 + 16 < p.N);

  __shared__ unsigned AsU[16 * ARS];        // 16 rows x 64 bf16 (32 dw) + pad
  __shared__ unsigned BsU[4][32 * BRS];     // per-wave: 32 n-rows x 64 bf16

  union { v8f v; float f[8]; } acc0, acc1;
#pragma unroll
  for (int i = 0; i < 8; ++i) { acc0.f[i] = 0.f; acc1.f[i] = 0.f; }

  const int mr = lane & 15, half = lane >> 4;

  for (int s = 0; s < p.nseg; ++s) {
    const Seg sg = p.seg[s];
    const bool xm = (p.aXmode != 0) && (s == 0);
    const unsigned short* Ab = xm ? nullptr : (sg.A + (long long)bz * sg.aBatch);
    const float* xb = xm ? (p.xsrc + (long long)bz * p.xBatch) : nullptr;

    for (int k0 = 0; k0 < sg.K; k0 += 64) {
      // ---- stage A tile (16 x 64 bf16) ----
      if (xm) {
        // transpose-on-load from f32 x: A[m][k] = x[(k0+k)*xRows + m]
        const int k  = tid >> 1;            // 0..63
        const int mq = (tid & 1) * 8;       // 8 m-values per thread
        unsigned short* arow = (unsigned short*)AsU;
#pragma unroll
        for (int j = 0; j < 8; ++j) {
          int m = m0 + mq + j;
          if (m >= p.xRows) m = p.xRows - 1;          // clamp ragged rows
          float v = xb[(long long)(k0 + k) * p.xRows + m];
          arow[(mq + j) * (2 * ARS) + k] = f2bf(v);
        }
      } else {
        const int row = tid >> 3;           // 16 rows, 8 threads/row, 16B each
        const int ch  = tid & 7;
        const unsigned short* gsrc = Ab + (long long)(m0 + row) * sg.lda + k0 + ch * 8;
        unsigned* ldst = &AsU[row * ARS + ch * 4];
#if HAS_ASYNC_LDS
        __builtin_amdgcn_global_load_async_to_lds_b128((g_v4i*)gsrc, (l_v4i*)ldst, 0, 0);
#else
        *(uint4*)ldst = *(const uint4*)gsrc;
#endif
      }
      // ---- stage B tile: 32 n-rows x 64 bf16, per wave ----
      if (n0 + lane < p.N) {
        const unsigned short* gsrc = sg.Bw + (long long)(n0 + lane) * sg.ldb + k0;
        unsigned* ldst = &BsU[wave][lane * BRS];
#pragma unroll
        for (int j = 0; j < 8; ++j) {
#if HAS_ASYNC_LDS
          __builtin_amdgcn_global_load_async_to_lds_b128(
              (g_v4i*)(gsrc + j * 8), (l_v4i*)(ldst + j * 4), 0, 0);
#else
          *(uint4*)(ldst + j * 4) = *(const uint4*)(gsrc + j * 8);
#endif
        }
      }
#if HAS_ASYNC_LDS
      __builtin_amdgcn_s_wait_asynccnt(0);
#endif
      __syncthreads();

      // ---- 2 k-subtiles x 2 n-tiles = 4 WMMAs per barrier pair ----
#pragma unroll
      for (int kt = 0; kt < 2; ++kt) {
        union { v16bf v; unsigned u[8]; } fa, fb;
        // A layout: dword v holds K = kt*32 + half*16 + 2v
#pragma unroll
        for (int v = 0; v < 8; ++v)
          fa.u[v] = AsU[mr * ARS + kt * 16 + half * 8 + v];
        if (act0) {
          // B layout: lane -> n = lane&15; dword v holds K = kt*32 + half*16 + 2v
#pragma unroll
          for (int v = 0; v < 8; ++v)
            fb.u[v] = BsU[wave][mr * BRS + kt * 16 + half * 8 + v];
          acc0.v = __builtin_amdgcn_wmma_f32_16x16x32_bf16(
              false, fa.v, false, fb.v, (short)0, acc0.v, false, false);
        }
        if (act1) {
#pragma unroll
          for (int v = 0; v < 8; ++v)
            fb.u[v] = BsU[wave][(16 + mr) * BRS + kt * 16 + half * 8 + v];
          acc1.v = __builtin_amdgcn_wmma_f32_16x16x32_bf16(
              false, fa.v, false, fb.v, (short)0, acc1.v, false, false);
        }
      }
      __syncthreads();
    }
  }

  // C/D layout: VGPR v -> m = v + 8*(lane>=16); n = lane&15
#pragma unroll
  for (int tile = 0; tile < 2; ++tile) {
    if (tile == 0 && !act0) continue;
    if (tile == 1 && !act1) continue;
    const float* accf = (tile == 0) ? acc0.f : acc1.f;
    const int gn = n0 + tile * 16 + mr;
    float bsum = 0.f;
    if (p.bias0) bsum += p.bias0[gn];
    if (p.bias1) bsum += p.bias1[gn];
#pragma unroll
    for (int v = 0; v < 8; ++v) {
      const int gm = m0 + half * 8 + v;
      if (gm >= p.Mstore) continue;
      float r = accf[v] + bsum;
      if (p.act == 1) r = tanhf(r);
      const long long off = (long long)bz * p.oBatch + (long long)gm * p.ldc + gn;
      if (p.outF) p.outF[off] = r;
      if (p.outB) p.outB[off] = f2bf(r);
    }
  }
}

// ------------------------------ small kernels ------------------------------
__global__ void convert_f32_bf16_kernel(const float* __restrict__ s,
                                        unsigned short* __restrict__ d, int n) {
  int i = blockIdx.x * blockDim.x + threadIdx.x;
  if (i < n) d[i] = f2bf(s[i]);
}

__global__ void zero_u32_kernel(unsigned* __restrict__ p, int n) {
  int i = blockIdx.x * blockDim.x + threadIdx.x;
  if (i < n) p[i] = 0u;
}

// AdaptiveMaxPool2d((1,1)): max over the 196 contiguous spatial elements.
__global__ void maxpool_kernel(const float* __restrict__ x,
                               unsigned short* __restrict__ out) {
  int i = blockIdx.x * blockDim.x + threadIdx.x;   // over B*ENC
  if (i >= B_ * ENC_) return;
  const float* r = x + (long long)i * HW_;
  float m = r[0];
  for (int k = 1; k < HW_; ++k) m = fmaxf(m, r[k]);
  out[i] = f2bf(m);
}

// BatchNorm1d (training batch stats) over B=128 per feature; writes t=0 slot.
__global__ void bn_kernel(const float* __restrict__ xf,
                          const float* __restrict__ g,
                          const float* __restrict__ bb,
                          unsigned short* __restrict__ inp0) {
  const int e = blockIdx.x;        // 512 features
  const int t = threadIdx.x;       // 128 = batch
  float v = xf[t * E_ + e];
  __shared__ float s1[128], s2[128];
  s1[t] = v; s2[t] = v * v;
  __syncthreads();
  for (int o = 64; o > 0; o >>= 1) {
    if (t < o) { s1[t] += s1[t + o]; s2[t] += s2[t + o]; }
    __syncthreads();
  }
  float mu  = s1[0] * (1.f / 128.f);
  float var = s2[0] * (1.f / 128.f) - mu * mu;
  float xbn = g[e] * (v - mu) * rsqrtf(var + 1e-5f) + bb[e];
  inp0[t * E_ + e] = f2bf(xbn);
}

// inputs[t][b][:] = emb[y[b][t-1]] for t = 1..31 (bf16)
__global__ void embed_kernel(const int* __restrict__ y,
                             const float* __restrict__ emb,
                             unsigned short* __restrict__ inp) {
  const int t = blockIdx.x + 1;
  const int b = blockIdx.y;
  const int tok = y[b * T_ + (t - 1)];
  const float* er = emb + (long long)tok * E_;
  unsigned short* dst = inp + ((long long)t * B_ + b) * E_;
  for (int e = threadIdx.x; e < E_; e += blockDim.x) dst[e] = f2bf(er[e]);
}

// Dot attention: scores = h0 . ctx ; softmax over 196 ; mix = wts @ ctx.
// One block per batch sample (M=1 matvec per sample -> VALU, not WMMA).
__global__ __launch_bounds__(256)
void attnprep_kernel(const unsigned short* __restrict__ h0b,
                     const unsigned short* __restrict__ ctx,
                     unsigned short* __restrict__ mixb) {
  const int b = blockIdx.x, tid = threadIdx.x;
  __shared__ float hs[E_];
  __shared__ float sc[HW_];
  __shared__ float red[256];
  for (int d = tid; d < E_; d += 256) hs[d] = bf2f(h0b[b * E_ + d]);
  __syncthreads();
  const unsigned short* cb = ctx + (long long)b * HW_ * D_;
  for (int k = tid; k < HW_; k += 256) {
    const unsigned short* row = cb + k * D_;
    float s = 0.f;
    for (int d = 0; d < D_; ++d) s += hs[d] * bf2f(row[d]);
    sc[k] = s;
  }
  __syncthreads();
  float m = -3.4e38f;
  for (int k = tid; k < HW_; k += 256) m = fmaxf(m, sc[k]);
  red[tid] = m;
  __syncthreads();
  for (int o = 128; o > 0; o >>= 1) {
    if (tid < o) red[tid] = fmaxf(red[tid], red[tid + o]);
    __syncthreads();
  }
  const float mx = red[0];
  __syncthreads();
  float ps = 0.f;
  for (int k = tid; k < HW_; k += 256) {
    float e = __expf(sc[k] - mx);
    sc[k] = e;
    ps += e;
  }
  red[tid] = ps;
  __syncthreads();
  for (int o = 128; o > 0; o >>= 1) {
    if (tid < o) red[tid] += red[tid + o];
    __syncthreads();
  }
  const float inv = 1.f / red[0];
  __syncthreads();
  for (int d = tid; d < D_; d += 256) {
    float a = 0.f;
    for (int k = 0; k < HW_; ++k) a += sc[k] * bf2f(cb[k * D_ + d]);
    mixb[b * D_ + d] = f2bf(a * inv);
  }
}

// LSTM pointwise: c' = sig(f)*c + sig(i)*tanh(g); h' = sig(o)*tanh(c')
__global__ void lstm_update_kernel(const float* __restrict__ gates,
                                   float* __restrict__ c,
                                   unsigned short* __restrict__ hb,
                                   unsigned short* __restrict__ outs, int t) {
  int i = blockIdx.x * blockDim.x + threadIdx.x;   // over B*D
  if (i >= B_ * D_) return;
  const int b = i >> 9, d = i & 511;
  const float* g = gates + (long long)b * 4 * D_;
  float ig = 1.f / (1.f + __expf(-g[d]));
  float fg = 1.f / (1.f + __expf(-g[D_ + d]));
  float gg = tanhf(g[2 * D_ + d]);
  float og = 1.f / (1.f + __expf(-g[3 * D_ + d]));
  float cn = fg * c[i] + ig * gg;
  c[i] = cn;
  float hn = og * tanhf(cn);
  hb[i] = f2bf(hn);
  if (outs) outs[((long long)b * T_ + t) * D_ + d] = f2bf(hn);
}

// ------------------------------- host side ---------------------------------
extern "C" void kernel_launch(void* const* d_in, const int* in_sizes, int n_in,
                              void* d_out, int out_size, void* d_ws,
                              size_t ws_size, hipStream_t stream) {
  (void)in_sizes; (void)n_in; (void)out_size; (void)ws_size;
  const float* x      = (const float*)d_in[0];
  const int*   y      = (const int*)d_in[1];
  // d_in[2] = lengths (== 32, compile-time constant here)
  const float* fc1_w  = (const float*)d_in[3];
  const float* fc1_b  = (const float*)d_in[4];
  const float* bn_g   = (const float*)d_in[5];
  const float* bn_b   = (const float*)d_in[6];
  const float* emb    = (const float*)d_in[7];
  const float* attn_w = (const float*)d_in[8];
  const float* attn_b = (const float*)d_in[9];
  const float* lin_w  = (const float*)d_in[10];
  const float* w_ih1  = (const float*)d_in[11];
  const float* w_hh1  = (const float*)d_in[12];
  const float* b_ih1  = (const float*)d_in[13];
  const float* b_hh1  = (const float*)d_in[14];
  const float* w_ih2  = (const float*)d_in[15];
  const float* w_hh2  = (const float*)d_in[16];
  const float* b_ih2  = (const float*)d_in[17];
  const float* b_hh2  = (const float*)d_in[18];
  const float* fc2_w  = (const float*)d_in[19];
  const float* fc2_b  = (const float*)d_in[20];
  float* out = (float*)d_out;

  char* w = (char*)d_ws;
  size_t off = 0;
  auto alloc = [&](size_t bytes) -> void* {
    void* p = w + off;
    off += (bytes + 255) & ~(size_t)255;
    return p;
  };
  unsigned short* fc1w_b  = (unsigned short*)alloc((size_t)E_ * ENC_ * 2);
  unsigned short* attnw_b = (unsigned short*)alloc((size_t)D_ * ENC_ * 2);
  unsigned short* linw_b  = (unsigned short*)alloc((size_t)D_ * 2 * D_ * 2);
  unsigned short* wih1_b  = (unsigned short*)alloc((size_t)4 * D_ * (E_ + D_) * 2);
  unsigned short* whh1_b  = (unsigned short*)alloc((size_t)4 * D_ * D_ * 2);
  unsigned short* wih2_b  = (unsigned short*)alloc((size_t)4 * D_ * D_ * 2);
  unsigned short* whh2_b  = (unsigned short*)alloc((size_t)4 * D_ * D_ * 2);
  unsigned short* fc2w_b  = (unsigned short*)alloc((size_t)V_ * D_ * 2);
  unsigned short* pooled_b= (unsigned short*)alloc((size_t)B_ * ENC_ * 2);
  float*          xf      = (float*)alloc((size_t)B_ * E_ * 4);
  unsigned short* inp     = (unsigned short*)alloc((size_t)T_ * B_ * E_ * 2);
  unsigned short* ctxb    = (unsigned short*)alloc((size_t)B_ * HW_ * D_ * 2);
  unsigned short* mixb    = (unsigned short*)alloc((size_t)B_ * D_ * 2);
  unsigned short* attnb   = (unsigned short*)alloc((size_t)B_ * D_ * 2);
  float*          gates   = (float*)alloc((size_t)B_ * 4 * D_ * 4);
  unsigned short* h0b     = (unsigned short*)alloc((size_t)B_ * D_ * 2);
  unsigned short* h1b     = (unsigned short*)alloc((size_t)B_ * D_ * 2);
  float*          c0      = (float*)alloc((size_t)B_ * D_ * 4);
  float*          c1      = (float*)alloc((size_t)B_ * D_ * 4);
  unsigned short* outsb   = (unsigned short*)alloc((size_t)B_ * T_ * D_ * 2);

  auto conv = [&](const float* s, unsigned short* d, int n) {
    convert_f32_bf16_kernel<<<(n + 255) / 256, 256, 0, stream>>>(s, d, n);
  };
  auto launchGemm = [&](GemmP& p, int Mpad, int batch) {
    dim3 g((p.N + 127) / 128, Mpad / 16, batch);
    wmma_gemm_kernel<<<g, dim3(BDIM), 0, stream>>>(p);
  };
  auto zero = [&](void* p, size_t bytes) {
    int n = (int)(bytes / 4);
    zero_u32_kernel<<<(n + 255) / 256, 256, 0, stream>>>((unsigned*)p, n);
  };

  // ---- one-time prep: bf16 weight copies, pooling, state init ----
  conv(fc1_w,  fc1w_b,  E_ * ENC_);
  conv(attn_w, attnw_b, D_ * ENC_);
  conv(lin_w,  linw_b,  D_ * 2 * D_);
  conv(w_ih1,  wih1_b,  4 * D_ * (E_ + D_));
  conv(w_hh1,  whh1_b,  4 * D_ * D_);
  conv(w_ih2,  wih2_b,  4 * D_ * D_);
  conv(w_hh2,  whh2_b,  4 * D_ * D_);
  conv(fc2_w,  fc2w_b,  V_ * D_);
  zero(h0b, (size_t)B_ * D_ * 2);
  zero(h1b, (size_t)B_ * D_ * 2);
  zero(c0,  (size_t)B_ * D_ * 4);
  zero(c1,  (size_t)B_ * D_ * 4);

  maxpool_kernel<<<(B_ * ENC_ + 255) / 256, 256, 0, stream>>>(x, pooled_b);

  { // fc1: xf = pooled @ fc1_w^T + fc1_b   (M=128, N=512, K=2048)
    GemmP p{};
    p.nseg = 1;
    p.seg[0] = {pooled_b, fc1w_b, 0, ENC_, ENC_, ENC_};
    p.bias0 = fc1_b; p.outF = xf; p.ldc = E_; p.N = E_; p.Mstore = B_;
    launchGemm(p, B_, 1);
  }
  bn_kernel<<<E_, B_, 0, stream>>>(xf, bn_g, bn_b, inp);       // inputs[0]
  embed_kernel<<<dim3(T_ - 1, B_), 256, 0, stream>>>(y, emb, inp);

  { // ctx[b] = x_[b] @ attn_w^T + attn_b  (batched: M=196 (pad 208), N=512, K=2048)
    GemmP p{};
    p.nseg = 1;
    p.aXmode = 1; p.xsrc = x; p.xBatch = (long long)ENC_ * HW_; p.xRows = HW_;
    p.seg[0] = {nullptr, attnw_b, 0, 0, ENC_, ENC_};
    p.bias0 = attn_b;
    p.outB = ctxb; p.oBatch = (long long)HW_ * D_; p.ldc = D_;
    p.N = D_; p.Mstore = HW_;
    launchGemm(p, 208, B_);
  }

  // ---- recurrence: 32 steps ----
  for (int t = 0; t < T_; ++t) {
    attnprep_kernel<<<B_, 256, 0, stream>>>(h0b, ctxb, mixb);

    { // attn = tanh([mix, h0] @ lin_out_w^T)   (M=128, N=512, K=1024 via 2 segs)
      GemmP p{};
      p.nseg = 2;
      p.seg[0] = {mixb, linw_b,        0, D_, 2 * D_, D_};
      p.seg[1] = {h0b,  linw_b + D_,   0, D_, 2 * D_, D_};
      p.outB = attnb; p.ldc = D_; p.N = D_; p.Mstore = B_; p.act = 1;
      launchGemm(p, B_, 1);
    }
    { // gates1 = [inp_t, attn]@w_ih1^T + h0@w_hh1^T + b_ih1 + b_hh1 (N=2048)
      GemmP p{};
      p.nseg = 3;
      p.seg[0] = {inp + (size_t)t * B_ * E_, wih1_b,      0, E_, E_ + D_, E_};
      p.seg[1] = {attnb,                     wih1_b + E_, 0, D_, E_ + D_, D_};
      p.seg[2] = {h0b,                       whh1_b,      0, D_, D_,      D_};
      p.bias0 = b_ih1; p.bias1 = b_hh1;
      p.outF = gates; p.ldc = 4 * D_; p.N = 4 * D_; p.Mstore = B_;
      launchGemm(p, B_, 1);
    }
    lstm_update_kernel<<<(B_ * D_ + 255) / 256, 256, 0, stream>>>(
        gates, c0, h0b, nullptr, t);

    { // gates2 = h0'@w_ih2^T + h1@w_hh2^T + b_ih2 + b_hh2 (N=2048)
      GemmP p{};
      p.nseg = 2;
      p.seg[0] = {h0b, wih2_b, 0, D_, D_, D_};
      p.seg[1] = {h1b, whh2_b, 0, D_, D_, D_};
      p.bias0 = b_ih2; p.bias1 = b_hh2;
      p.outF = gates; p.ldc = 4 * D_; p.N = 4 * D_; p.Mstore = B_;
      launchGemm(p, B_, 1);
    }
    lstm_update_kernel<<<(B_ * D_ + 255) / 256, 256, 0, stream>>>(
        gates, c1, h1b, outsb, t);
  }

  { // logits = outs @ fc2_w^T + fc2_b   (M=4096, N=10000, K=512)
    GemmP p{};
    p.nseg = 1;
    p.seg[0] = {outsb, fc2w_b, 0, D_, D_, D_};
    p.bias0 = fc2_b; p.outF = out; p.ldc = V_; p.N = V_; p.Mstore = B_ * T_;
    launchGemm(p, B_ * T_, 1);
  }
}
